// group_decoder_sf_66864050864857
// MI455X (gfx1250) — compile-verified, matrix operand order checked by reading, not run on hardware
//
#include <hip/hip_runtime.h>
#include <hip/hip_bf16.h>

// ---------------------------------------------------------------------------
// Types for CDNA5 WMMA
// ---------------------------------------------------------------------------
typedef __attribute__((ext_vector_type(16))) __bf16 v16bf;
typedef __attribute__((ext_vector_type(8)))  float  v8f;

union FragAB {
    v16bf v;
    uint4 q[2];
};

// ---------------------------------------------------------------------------
// Helpers
// ---------------------------------------------------------------------------
__device__ __forceinline__ unsigned short f2bf(float f) {
    unsigned int u = __float_as_uint(f);
    unsigned int r = u + 0x7fffu + ((u >> 16) & 1u);   // round-to-nearest-even
    return (unsigned short)(r >> 16);
}
__device__ __forceinline__ float bf2f(unsigned short h) {
    return __uint_as_float(((unsigned int)h) << 16);
}
__device__ __forceinline__ float gelu_f(float x) {
    return 0.5f * x * (1.0f + erff(x * 0.7071067811865475f));
}

static constexpr int BB = 16;     // batch
static constexpr int DD = 512;    // channels
static constexpr int NN = 4096;   // positions

// ---------------------------------------------------------------------------
// Kernel: group-token pipeline (mlp_s -> mlp_c -> k/v projections).
// One block per batch. All tiny GEMV work kept in LDS.
// ---------------------------------------------------------------------------
__global__ __launch_bounds__(256) void k_token(
    const float* __restrict__ gtok,                          // (B, 512, 2)
    const float* __restrict__ ws1_w, const float* __restrict__ ws1_b,
    const float* __restrict__ ws2_w, const float* __restrict__ ws2_b,
    const float* __restrict__ wc1_w, const float* __restrict__ wc1_b,
    const float* __restrict__ wc2_w, const float* __restrict__ wc2_b,
    const float* __restrict__ kw, const float* __restrict__ kb,
    const float* __restrict__ vw, const float* __restrict__ vb,
    float* __restrict__ kout, float* __restrict__ vout)      // (B, 512, 2)
{
    const int b   = blockIdx.x;
    const int tid = threadIdx.x;
    __shared__ float gt1[512][2];
    __shared__ float hid[2048][2];
    __shared__ float gt2[512][2];

    // stage 1: mlp_s over positions d (channels of gtok): 2 -> 256 -> gelu -> 2
    for (int d = tid; d < 512; d += 256) {
        float a0 = gtok[(size_t)b * 1024 + d * 2 + 0];
        float a1 = gtok[(size_t)b * 1024 + d * 2 + 1];
        float s0 = ws2_b[0], s1 = ws2_b[1];
        for (int o = 0; o < 256; ++o) {
            float h = gelu_f(ws1_w[o * 2 + 0] * a0 + ws1_w[o * 2 + 1] * a1 + ws1_b[o]);
            s0 += ws2_w[o] * h;
            s1 += ws2_w[256 + o] * h;
        }
        gt1[d][0] = a0 + s0;
        gt1[d][1] = a1 + s1;
    }
    __syncthreads();

    // stage 2: mlp_c hidden 512 -> 2048, gelu
    for (int o = tid; o < 2048; o += 256) {
        float acc0 = wc1_b[o], acc1 = wc1_b[o];
        const float* wr = wc1_w + (size_t)o * 512;
        for (int i = 0; i < 512; ++i) {
            float w = wr[i];
            acc0 += w * gt1[i][0];
            acc1 += w * gt1[i][1];
        }
        hid[o][0] = gelu_f(acc0);
        hid[o][1] = gelu_f(acc1);
    }
    __syncthreads();

    // stage 3: 2048 -> 512, residual
    for (int d = tid; d < 512; d += 256) {
        float acc0 = wc2_b[d], acc1 = wc2_b[d];
        const float* wr = wc2_w + (size_t)d * 2048;
        for (int i = 0; i < 2048; ++i) {
            float w = wr[i];
            acc0 += w * hid[i][0];
            acc1 += w * hid[i][1];
        }
        gt2[d][0] = gt1[d][0] + acc0;
        gt2[d][1] = gt1[d][1] + acc1;
    }
    __syncthreads();

    // stage 4: k/v projections (512x512 each)
    for (int o = tid; o < 512; o += 256) {
        float k0 = kb[o], k1 = kb[o], v0 = vb[o], v1 = vb[o];
        const float* kr = kw + (size_t)o * 512;
        const float* vr = vw + (size_t)o * 512;
        for (int i = 0; i < 512; ++i) {
            float g0 = gt2[i][0], g1 = gt2[i][1];
            k0 += kr[i] * g0; k1 += kr[i] * g1;
            v0 += vr[i] * g0; v1 += vr[i] * g1;
        }
        kout[((size_t)b * 512 + o) * 2 + 0] = k0;
        kout[((size_t)b * 512 + o) * 2 + 1] = k1;
        vout[((size_t)b * 512 + o) * 2 + 0] = v0;
        vout[((size_t)b * 512 + o) * 2 + 1] = v1;
    }
}

// ---------------------------------------------------------------------------
// f32 -> bf16 conversions
// ---------------------------------------------------------------------------
__global__ void k_f2bf(const float* __restrict__ in, unsigned short* __restrict__ out,
                       long long n) {
    long long i = (long long)blockIdx.x * blockDim.x + threadIdx.x;
    long long stride = (long long)gridDim.x * blockDim.x;
    for (; i < n; i += stride) out[i] = f2bf(in[i]);
}

// x (B,512,N) f32 -> first 512 rows of h buffer (B,1024,N) bf16 (free concat)
__global__ void k_x2h(const float* __restrict__ x, unsigned short* __restrict__ h) {
    long long total = (long long)BB * DD * NN;
    long long i = (long long)blockIdx.x * blockDim.x + threadIdx.x;
    long long stride = (long long)gridDim.x * blockDim.x;
    for (; i < total; i += stride) {
        long long b = i / ((long long)DD * NN);
        h[i + b * (long long)DD * NN] = f2bf(x[i]);
    }
}

// ---------------------------------------------------------------------------
// Generic batched bf16 WMMA GEMM:  C[z] = W (MxK) * X[z] (KxN) + bias
//   block = 256 threads = 8 waves arranged 4(M) x 2(N); wave tile 32x64
//   -> block tile 128x128, 8 v_wmma_f32_16x16x32_bf16 per wave per K-step.
//   A fragments loaded straight from global (K-contiguous, 2x b128 per lane).
//   X tile staged into LDS transposed (packed b32 stores) -> B fragments are
//   2x ds_load_b128 each.
// mode: 0 = store bf16, 1 = store bf16(gelu), 2 = store f32
// ---------------------------------------------------------------------------
__global__ __launch_bounds__(256) void k_gemm(
    const unsigned short* __restrict__ W,
    const unsigned short* __restrict__ X,
    const float* __restrict__ bias,
    void* __restrict__ Out,
    int M, int N, int K,
    long long xBatchStride, long long oBatchStride,
    int mode)
{
    __shared__ __align__(16) unsigned short ldsXT[128][40];  // [n_local][k], padded

    const int tid  = threadIdx.x;
    const int lane = tid & 31;
    const int wave = tid >> 5;
    const int wm   = wave >> 1;          // 0..3  (M direction)
    const int wn   = wave & 1;           // 0..1  (N direction)
    const int z    = blockIdx.z;
    const int nblk = blockIdx.x * 128;
    const int mblk = blockIdx.y * 128;

    const unsigned short* Xb = X + (size_t)z * xBatchStride;

    const int l  = lane & 15;
    const int kh = (lane < 16) ? 0 : 8;  // K-half select per ISA 16-bit layout
    const int mr0 = mblk + wm * 32;
    const int nc0 = nblk + wn * 64;

    // clamp A rows for the M < blockM case (g1: M = 64); stores are predicated
    const int mA0 = (mr0 + l      < M) ? (mr0 + l)      : 0;
    const int mA1 = (mr0 + 16 + l < M) ? (mr0 + 16 + l) : 0;

    v8f acc[2][4];
#pragma unroll
    for (int a = 0; a < 2; ++a)
#pragma unroll
        for (int c = 0; c < 4; ++c)
#pragma unroll
            for (int i = 0; i < 8; ++i) acc[a][c][i] = 0.f;

    for (int kb = 0; kb < K; kb += 32) {
        __syncthreads();
        // cooperative load of X tile (32 K-rows x 128 cols), transposed into
        // LDS with packed 2xK b32 stores. chunk = 2k x 4n (8 elems); 2 / thread
#pragma unroll
        for (int it = 0; it < 2; ++it) {
            int c  = tid + it * 256;          // 0..511
            int kr = (c >> 5) << 1;           // 0,2,..,30
            int nc = (c & 31) << 2;           // 0..124
            const unsigned short* src = Xb + (size_t)(kb + kr) * N + nblk + nc;
            __builtin_prefetch(src + (size_t)32 * N, 0, 3);
            uint2 d0 = *(const uint2*)src;                  // row k,   n..n+3
            uint2 d1 = *(const uint2*)(src + N);            // row k+1, n..n+3
            unsigned int p0 = (d0.x & 0xffffu) | (d1.x << 16);            // n+0
            unsigned int p1 = (d0.x >> 16)     | (d1.x & 0xffff0000u);    // n+1
            unsigned int p2 = (d0.y & 0xffffu) | (d1.y << 16);            // n+2
            unsigned int p3 = (d0.y >> 16)     | (d1.y & 0xffff0000u);    // n+3
            *(unsigned int*)&ldsXT[nc + 0][kr] = p0;
            *(unsigned int*)&ldsXT[nc + 1][kr] = p1;
            *(unsigned int*)&ldsXT[nc + 2][kr] = p2;
            *(unsigned int*)&ldsXT[nc + 3][kr] = p3;
        }
        __syncthreads();

        FragAB fa[2], fb[4];
        const unsigned short* pa0 = W + (size_t)mA0 * K + kb + kh;
        const unsigned short* pa1 = W + (size_t)mA1 * K + kb + kh;
        fa[0].q[0] = *(const uint4*)pa0;   fa[0].q[1] = *(const uint4*)(pa0 + 16);
        fa[1].q[0] = *(const uint4*)pa1;   fa[1].q[1] = *(const uint4*)(pa1 + 16);
#pragma unroll
        for (int c = 0; c < 4; ++c) {
            const unsigned short* pb = &ldsXT[wn * 64 + c * 16 + l][kh];
            fb[c].q[0] = *(const uint4*)pb;
            fb[c].q[1] = *(const uint4*)(pb + 16);
        }
#pragma unroll
        for (int a = 0; a < 2; ++a)
#pragma unroll
            for (int c = 0; c < 4; ++c)
                acc[a][c] = __builtin_amdgcn_wmma_f32_16x16x32_bf16(
                    false, fa[a].v, false, fb[c].v, (short)0, acc[a][c], false, false);
    }

    // epilogue: C layout — VGPR j: lanes 0-15 -> M=j, lanes 16-31 -> M=j+8
    const int rowoff = (lane < 16) ? 0 : 8;
#pragma unroll
    for (int a = 0; a < 2; ++a) {
#pragma unroll
        for (int c = 0; c < 4; ++c) {
            const int mt = mr0 + a * 16;
            const int nt = nc0 + c * 16;
#pragma unroll
            for (int j = 0; j < 8; ++j) {
                int row = mt + rowoff + j;
                int col = nt + l;
                if (row < M) {
                    float v = acc[a][c][j] + bias[row];
                    if (mode == 1) v = gelu_f(v);
                    size_t idx = (size_t)z * oBatchStride + (size_t)row * N + col;
                    if (mode == 2) ((float*)Out)[idx] = v;
                    else           ((unsigned short*)Out)[idx] = f2bf(v);
                }
            }
        }
    }
}

// ---------------------------------------------------------------------------
// Attention over 2 group tokens: softmax over m in {0,1}, 4 heads of 128 dims,
// channel c maps to head h = c & 3 (reshape (b, HD, H, N)).
// ---------------------------------------------------------------------------
__global__ __launch_bounds__(256) void k_attn(
    const unsigned short* __restrict__ q,     // (B,512,N) bf16
    const float* __restrict__ kbuf,           // (B,512,2)
    const float* __restrict__ vbuf,           // (B,512,2)
    unsigned short* __restrict__ out)         // (B,512,N) bf16
{
    const int b = blockIdx.z;
    const int n = blockIdx.x * 256 + threadIdx.x;
    __shared__ float kk[512][2];
    __shared__ float vv[512][2];
    for (int c = threadIdx.x; c < 512; c += 256) {
        kk[c][0] = kbuf[((size_t)b * 512 + c) * 2 + 0];
        kk[c][1] = kbuf[((size_t)b * 512 + c) * 2 + 1];
        vv[c][0] = vbuf[((size_t)b * 512 + c) * 2 + 0];
        vv[c][1] = vbuf[((size_t)b * 512 + c) * 2 + 1];
    }
    __syncthreads();

    const unsigned short* qb = q + (size_t)b * 512 * NN + n;
    float lg[4][2] = {};
    for (int c = 0; c < 512; ++c) {
        float qv = bf2f(qb[(size_t)c * NN]);
        int h = c & 3;
        lg[h][0] += qv * kk[c][0];
        lg[h][1] += qv * kk[c][1];
    }
    const float sc = 0.08838834764831845f;   // 128^-0.5
    float a[4][2];
#pragma unroll
    for (int h = 0; h < 4; ++h) {
        float x0 = lg[h][0] * sc, x1 = lg[h][1] * sc;
        float m = fmaxf(x0, x1);
        float e0 = __expf(x0 - m), e1 = __expf(x1 - m);
        float s = e0 + e1;
        a[h][0] = e0 / s; a[h][1] = e1 / s;
    }
    unsigned short* ob = out + (size_t)b * 512 * NN + n;
    for (int c = 0; c < 512; ++c) {
        int h = c & 3;
        ob[(size_t)c * NN] = f2bf(a[h][0] * vv[c][0] + a[h][1] * vv[c][1]);
    }
}

// ---------------------------------------------------------------------------
// Channel LayerNorm of merged (f32) -> bf16 into rows [512..1023] of h buffer
// ---------------------------------------------------------------------------
__global__ __launch_bounds__(256) void k_ln_h(
    const float* __restrict__ merged, const float* __restrict__ g,
    const float* __restrict__ bt, unsigned short* __restrict__ hbuf)
{
    const int b = blockIdx.z;
    const int n = blockIdx.x * 256 + threadIdx.x;
    const float* p = merged + (size_t)b * 512 * NN + n;
    float s = 0.f, ss = 0.f;
    for (int c = 0; c < 512; ++c) { float v = p[(size_t)c * NN]; s += v; ss += v * v; }
    float m = s * (1.f / 512.f);
    float var = ss * (1.f / 512.f) - m * m;
    float r = rsqrtf(var + 1e-5f);
    unsigned short* o = hbuf + ((size_t)b * 1024 + 512) * NN + n;
    for (int c = 0; c < 512; ++c)
        o[(size_t)c * NN] = f2bf((p[(size_t)c * NN] - m) * r * g[c] + bt[c]);
}

// xd = x + LN(m2out) -> bf16
__global__ __launch_bounds__(256) void k_ln_xd(
    const float* __restrict__ m2o, const float* __restrict__ x,
    const float* __restrict__ g, const float* __restrict__ bt,
    unsigned short* __restrict__ xd)
{
    const int b = blockIdx.z;
    const int n = blockIdx.x * 256 + threadIdx.x;
    const float* p  = m2o + (size_t)b * 512 * NN + n;
    const float* xp = x   + (size_t)b * 512 * NN + n;
    float s = 0.f, ss = 0.f;
    for (int c = 0; c < 512; ++c) { float v = p[(size_t)c * NN]; s += v; ss += v * v; }
    float m = s * (1.f / 512.f);
    float var = ss * (1.f / 512.f) - m * m;
    float r = rsqrtf(var + 1e-5f);
    unsigned short* o = xd + (size_t)b * 512 * NN + n;
    for (int c = 0; c < 512; ++c) {
        float v = xp[(size_t)c * NN] + ((p[(size_t)c * NN] - m) * r * g[c] + bt[c]);
        o[(size_t)c * NN] = f2bf(v);
    }
}

// ---------------------------------------------------------------------------
// Grouping tail: LN64+gelu -> 16 -> LN+gelu -> 4 -> LN+gelu -> 2 -> softmax
// ---------------------------------------------------------------------------
__global__ __launch_bounds__(256) void k_tail(
    const float* __restrict__ g1o,
    const float* __restrict__ l1g, const float* __restrict__ l1b,
    const float* __restrict__ w2,  const float* __restrict__ b2,
    const float* __restrict__ l2g, const float* __restrict__ l2b,
    const float* __restrict__ w3,  const float* __restrict__ b3,
    const float* __restrict__ l3g, const float* __restrict__ l3b,
    const float* __restrict__ w4,  float* __restrict__ outp)
{
    const int b = blockIdx.z;
    const int n = blockIdx.x * 256 + threadIdx.x;
    const float* p = g1o + (size_t)b * 64 * NN + n;

    float v[64]; float s = 0.f;
#pragma unroll
    for (int c = 0; c < 64; ++c) { v[c] = p[(size_t)c * NN]; s += v[c]; }
    float m = s * (1.f / 64.f), ss = 0.f;
#pragma unroll
    for (int c = 0; c < 64; ++c) { float d = v[c] - m; ss += d * d; }
    float r = rsqrtf(ss * (1.f / 64.f) + 1e-5f);
#pragma unroll
    for (int c = 0; c < 64; ++c) v[c] = gelu_f((v[c] - m) * r * l1g[c] + l1b[c]);

    float u[16]; s = 0.f;
#pragma unroll
    for (int o = 0; o < 16; ++o) {
        float a = b2[o];
#pragma unroll
        for (int c = 0; c < 64; ++c) a += w2[o * 64 + c] * v[c];
        u[o] = a; s += a;
    }
    m = s * (1.f / 16.f); ss = 0.f;
#pragma unroll
    for (int o = 0; o < 16; ++o) { float d = u[o] - m; ss += d * d; }
    r = rsqrtf(ss * (1.f / 16.f) + 1e-5f);
#pragma unroll
    for (int o = 0; o < 16; ++o) u[o] = gelu_f((u[o] - m) * r * l2g[o] + l2b[o]);

    float t[4]; s = 0.f;
#pragma unroll
    for (int o = 0; o < 4; ++o) {
        float a = b3[o];
#pragma unroll
        for (int c = 0; c < 16; ++c) a += w3[o * 16 + c] * u[c];
        t[o] = a; s += a;
    }
    m = s * 0.25f; ss = 0.f;
#pragma unroll
    for (int o = 0; o < 4; ++o) { float d = t[o] - m; ss += d * d; }
    r = rsqrtf(ss * 0.25f + 1e-5f);
#pragma unroll
    for (int o = 0; o < 4; ++o) t[o] = gelu_f((t[o] - m) * r * l3g[o] + l3b[o]);

    float z0 = w4[0] * t[0] + w4[1] * t[1] + w4[2] * t[2] + w4[3] * t[3];
    float z1 = w4[4] * t[0] + w4[5] * t[1] + w4[6] * t[2] + w4[7] * t[3];
    float mx = fmaxf(z0, z1);
    float e0 = __expf(z0 - mx), e1 = __expf(z1 - mx);
    outp[(size_t)b * NN + n] = e0 / (e0 + e1);
}

// ---------------------------------------------------------------------------
// Launch
// ---------------------------------------------------------------------------
extern "C" void kernel_launch(void* const* d_in, const int* in_sizes, int n_in,
                              void* d_out, int out_size, void* d_ws, size_t ws_size,
                              hipStream_t stream) {
    const float* gtok  = (const float*)d_in[0];
    const float* x     = (const float*)d_in[1];
    const float* ws1w  = (const float*)d_in[2];  const float* ws1b = (const float*)d_in[3];
    const float* ws2w  = (const float*)d_in[4];  const float* ws2b = (const float*)d_in[5];
    const float* wc1w  = (const float*)d_in[6];  const float* wc1b = (const float*)d_in[7];
    const float* wc2w  = (const float*)d_in[8];  const float* wc2b = (const float*)d_in[9];
    const float* qw    = (const float*)d_in[10]; const float* qb   = (const float*)d_in[11];
    const float* kw    = (const float*)d_in[12]; const float* kb   = (const float*)d_in[13];
    const float* vw    = (const float*)d_in[14]; const float* vb   = (const float*)d_in[15];
    const float* mgw   = (const float*)d_in[16]; const float* mgb  = (const float*)d_in[17];
    const float* m1w   = (const float*)d_in[18]; const float* m1b  = (const float*)d_in[19];
    const float* m2w   = (const float*)d_in[20]; const float* m2b  = (const float*)d_in[21];
    const float* n0g   = (const float*)d_in[22]; const float* n0b  = (const float*)d_in[23];
    const float* n1g   = (const float*)d_in[24]; const float* n1b  = (const float*)d_in[25];
    const float* g1w   = (const float*)d_in[26]; const float* g1b  = (const float*)d_in[27];
    const float* gl1g  = (const float*)d_in[28]; const float* gl1b = (const float*)d_in[29];
    const float* g2w   = (const float*)d_in[30]; const float* g2b  = (const float*)d_in[31];
    const float* gl2g  = (const float*)d_in[32]; const float* gl2b = (const float*)d_in[33];
    const float* g3w   = (const float*)d_in[34]; const float* g3b  = (const float*)d_in[35];
    const float* gl3g  = (const float*)d_in[36]; const float* gl3b = (const float*)d_in[37];
    const float* g4w   = (const float*)d_in[38];

    // workspace carve-out (256B aligned slices)
    char* base = (char*)d_ws;
    size_t off = 0;
    auto alloc = [&](size_t bytes) -> void* {
        void* p = base + off;
        off += (bytes + 255) & ~(size_t)255;
        return p;
    };
    float*          kbuf = (float*)alloc((size_t)BB * 512 * 2 * 4);
    float*          vbuf = (float*)alloc((size_t)BB * 512 * 2 * 4);
    unsigned short* wq   = (unsigned short*)alloc((size_t)512 * 512 * 2);
    unsigned short* wmg  = (unsigned short*)alloc((size_t)512 * 512 * 2);
    unsigned short* wm1  = (unsigned short*)alloc((size_t)1024 * 1024 * 2);
    unsigned short* wm2  = (unsigned short*)alloc((size_t)512 * 1024 * 2);
    unsigned short* wg1  = (unsigned short*)alloc((size_t)64 * 512 * 2);
    float*          bufA = (float*)alloc((size_t)BB * 512 * NN * 4);          // merged, then m2out
    unsigned short* hbuf = (unsigned short*)alloc((size_t)BB * 1024 * NN * 2); // [x ; LN(merged)]
    unsigned short* bufC = (unsigned short*)alloc((size_t)BB * 1024 * NN * 2); // q/attn -> m1out -> xd
    float*          g1o  = (float*)alloc((size_t)BB * 64 * NN * 4);

    unsigned short* qbuf = bufC;
    unsigned short* attn = bufC + (size_t)BB * 512 * NN;
    unsigned short* m1o  = bufC;                // overlays q+attn (dead by then)
    unsigned short* xd   = bufC;                // overlays m1out (dead by then)

    // 1) token pipeline -> k, v
    k_token<<<BB, 256, 0, stream>>>(gtok, ws1w, ws1b, ws2w, ws2b, wc1w, wc1b,
                                    wc2w, wc2b, kw, kb, vw, vb, kbuf, vbuf);

    // 2) weight conversions + x -> h (rows 0..511)
    k_f2bf<<<512, 256, 0, stream>>>(qw,  wq,  (long long)512 * 512);
    k_f2bf<<<512, 256, 0, stream>>>(mgw, wmg, (long long)512 * 512);
    k_f2bf<<<2048, 256, 0, stream>>>(m1w, wm1, (long long)1024 * 1024);
    k_f2bf<<<1024, 256, 0, stream>>>(m2w, wm2, (long long)512 * 1024);
    k_f2bf<<<128, 256, 0, stream>>>(g1w, wg1, (long long)64 * 512);
    k_x2h<<<8192, 256, 0, stream>>>(x, hbuf);

    const dim3 blk(256);
    // 3) q = qw @ x + qb  (bf16 out)
    k_gemm<<<dim3(NN / 128, 512 / 128, BB), blk, 0, stream>>>(
        wq, hbuf, qb, qbuf, 512, NN, 512,
        (long long)1024 * NN, (long long)512 * NN, 0);

    // 4) attention (2-token softmax)
    k_attn<<<dim3(NN / 256, 1, BB), blk, 0, stream>>>(qbuf, kbuf, vbuf, attn);

    // 5) merged = mgw @ attn + mgb  (f32 out)
    k_gemm<<<dim3(NN / 128, 512 / 128, BB), blk, 0, stream>>>(
        wmg, attn, mgb, bufA, 512, NN, 512,
        (long long)512 * NN, (long long)512 * NN, 2);

    // 6) h rows 512..1023 = bf16(LN(merged))
    k_ln_h<<<dim3(NN / 256, 1, BB), blk, 0, stream>>>(bufA, n0g, n0b, hbuf);

    // 7) m1out = gelu(m1w @ h + m1b)  (bf16 out)
    k_gemm<<<dim3(NN / 128, 1024 / 128, BB), blk, 0, stream>>>(
        wm1, hbuf, m1b, m1o, 1024, NN, 1024,
        (long long)1024 * NN, (long long)1024 * NN, 1);

    // 8) m2out = m2w @ m1out + m2b  (f32 out)
    k_gemm<<<dim3(NN / 128, 512 / 128, BB), blk, 0, stream>>>(
        wm2, m1o, m2b, bufA, 512, NN, 1024,
        (long long)1024 * NN, (long long)512 * NN, 2);

    // 9) xd = bf16(x + LN(m2out))
    k_ln_xd<<<dim3(NN / 256, 1, BB), blk, 0, stream>>>(bufA, x, n1g, n1b, xd);

    // 10) g1out = g1w @ xd + g1b  (f32 out; M=64 handled by row predication)
    k_gemm<<<dim3(NN / 128, 1, BB), blk, 0, stream>>>(
        wg1, xd, g1b, g1o, 64, NN, 512,
        (long long)512 * NN, (long long)64 * NN, 2);

    // 11) grouping tail -> softmax channel 0
    k_tail<<<dim3(NN / 256, 1, BB), blk, 0, stream>>>(
        g1o, gl1g, gl1b, g2w, g2b, gl2g, gl2b, g3w, g3b, gl3g, gl3b, g4w,
        (float*)d_out);
}
